// GPT_52819507806684
// MI455X (gfx1250) — compile-verified, matrix operand order checked by reading, not run on hardware
//
#include <hip/hip_runtime.h>
#include <hip/hip_bf16.h>

// ---------------------------------------------------------------------------
// GPT forward for gfx1250 (CDNA5, wave32, WMMA 16x16x32 f16 / f32-acc)
// ---------------------------------------------------------------------------

typedef __attribute__((ext_vector_type(16))) _Float16 v16h;
typedef __attribute__((ext_vector_type(8)))  _Float16 v8h;
typedef __attribute__((ext_vector_type(8)))  float    v8f;

#define DEV static __device__ __forceinline__

DEV v8f vzero8() {
    v8f z;
#pragma unroll
    for (int i = 0; i < 8; ++i) z[i] = 0.0f;
    return z;
}

// Convert 8 consecutive f32 to a packed v8h (two float4 loads -> 16B of f16).
DEV v8h cvt8(const float* g) {
    float4 f0 = *(const float4*)g;
    float4 f1 = *(const float4*)(g + 4);
    v8h r;
    r[0] = (_Float16)f0.x; r[1] = (_Float16)f0.y;
    r[2] = (_Float16)f0.z; r[3] = (_Float16)f0.w;
    r[4] = (_Float16)f1.x; r[5] = (_Float16)f1.y;
    r[6] = (_Float16)f1.z; r[7] = (_Float16)f1.w;
    return r;
}

// Assemble a 16x32 f16 WMMA A/B fragment from two 16-byte LDS chunks.
// Layout (ISA 7.12.2, 16-bit A 16x32): lanes 0-15 row=lane, K=0..7 then 16..23;
// lanes 16-31 row=lane-16, K=8..15 then 24..31.  Chunks are contiguous halves.
DEV v16h ld_frag(const _Float16* lo, const _Float16* hi) {
    v8h a = *(const v8h*)lo;
    v8h b = *(const v8h*)hi;
    return __builtin_shufflevector(a, b, 0, 1, 2, 3, 4, 5, 6, 7,
                                         8, 9, 10, 11, 12, 13, 14, 15);
}

DEV v8f wmma_f16(v16h a, v16h b, v8f c) {
    return __builtin_amdgcn_wmma_f32_16x16x32_f16(
        /*neg_a=*/false, a, /*neg_b=*/false, b,
        /*c_mod=*/(short)0, c, /*reuse_a=*/false, /*reuse_b=*/false);
}

// ---------------------------------------------------------------------------
// Weight convert + transpose:  in f32 [K][N]  ->  out f16 [N][K]
// ---------------------------------------------------------------------------
__global__ __launch_bounds__(256) void wconv_kernel(
    const float* __restrict__ in, _Float16* __restrict__ out, int K, int N) {
    __shared__ float tile[32][33];
    const int n0 = blockIdx.x * 32, k0 = blockIdx.y * 32;
    const int tx = threadIdx.x, ty = threadIdx.y;
#pragma unroll
    for (int i = 0; i < 4; ++i)
        tile[ty + i * 8][tx] = in[(size_t)(k0 + ty + i * 8) * N + n0 + tx];
    __syncthreads();
#pragma unroll
    for (int i = 0; i < 4; ++i)
        out[(size_t)(n0 + ty + i * 8) * K + k0 + tx] =
            (_Float16)tile[tx][ty + i * 8];
}

// Elementwise convert f32 -> f16 (head_w is already [V][D] = [N][K])
__global__ __launch_bounds__(256) void hconv_kernel(
    const float* __restrict__ in, _Float16* __restrict__ out, int n) {
    int i = blockIdx.x * 256 + threadIdx.x;
    if (i < n) out[i] = (_Float16)in[i];
}

// ---------------------------------------------------------------------------
// Embedding: x[bt][d] = tok_emb[idx[bt]][d] + pos_emb[t][d]
// ---------------------------------------------------------------------------
__global__ __launch_bounds__(256) void embed_kernel(
    const int* __restrict__ idx, const float* __restrict__ tok,
    const float* __restrict__ pos, float* __restrict__ X, int T, int D) {
    const int bt = blockIdx.x;
    const int t = bt % T;
    const int token = idx[bt];
    for (int i = threadIdx.x; i < D; i += 256)
        X[(size_t)bt * D + i] = tok[(size_t)token * D + i] + pos[(size_t)t * D + i];
}

// ---------------------------------------------------------------------------
// LayerNorm: one 256-thread block per row
// ---------------------------------------------------------------------------
__global__ __launch_bounds__(256) void ln_kernel(
    const float* __restrict__ X, const float* __restrict__ w,
    const float* __restrict__ b, float* __restrict__ Out, int D) {
    __shared__ float s_sum[8], s_sq[8], s_stats[2];
    const int t = threadIdx.x, lane = t & 31, wid = t >> 5;
    const float* x = X + (size_t)blockIdx.x * D;
    float s = 0.0f, ss = 0.0f;
    for (int i = t; i < D; i += 256) {
        float v = x[i];
        s += v;
        ss += v * v;
    }
#pragma unroll
    for (int m = 16; m >= 1; m >>= 1) {
        s += __shfl_xor(s, m, 32);
        ss += __shfl_xor(ss, m, 32);
    }
    if (lane == 0) { s_sum[wid] = s; s_sq[wid] = ss; }
    __syncthreads();
    if (t == 0) {
        float a = 0.0f, q = 0.0f;
#pragma unroll
        for (int i = 0; i < 8; ++i) { a += s_sum[i]; q += s_sq[i]; }
        float mu = a / (float)D;
        float var = q / (float)D - mu * mu;
        s_stats[0] = mu;
        s_stats[1] = rsqrtf(var + 1e-5f);
    }
    __syncthreads();
    const float mu = s_stats[0], rstd = s_stats[1];
    float* o = Out + (size_t)blockIdx.x * D;
    for (int i = t; i < D; i += 256)
        o[i] = (x[i] - mu) * rstd * w[i] + b[i];
}

// ---------------------------------------------------------------------------
// WMMA GEMM:  C[M][N] = A(f32,[M][K]) * Bt(f16,[N][K])^T  (+bias,+gelu,+res)
// 256 threads (8 waves), 128x128 block tile, BK=32, wave tile 64x32 (4x2 WMMA)
// Register double-buffering: global loads for slab k+1 issued before the
// WMMAs of slab k so HBM latency hides under the matrix pipe.
// ---------------------------------------------------------------------------
template <bool HAS_BIAS, bool DO_GELU, bool DO_RES>
__global__ __launch_bounds__(256) void gemm_kernel(
    const float* __restrict__ A, const _Float16* __restrict__ Bt,
    const float* __restrict__ bias, const float* __restrict__ res,
    float* __restrict__ C, int M, int N, int K) {
    __shared__ _Float16 sa[128 * 40];  // [m-row][k], row stride 40 (80B, 16B-aligned)
    __shared__ _Float16 sb[128 * 40];  // [n-row][k]

    const int t = threadIdx.x;
    const int lane = t & 31, wid = t >> 5;
    const int wm = wid >> 2, wn = wid & 3;  // 2x4 wave grid
    const int m0 = blockIdx.y * 128, n0 = blockIdx.x * 128;
    const int l16 = lane & 15;
    const int hs = lane >> 4;          // half-wave select
    const int kb = hs ? 8 : 0;         // K base per WMMA 16-bit layout

    v8f acc[4][2];
#pragma unroll
    for (int mi = 0; mi < 4; ++mi)
#pragma unroll
        for (int ni = 0; ni < 2; ++ni) acc[mi][ni] = vzero8();

    const int lrow = t >> 1;
    const int lseg = (t & 1) * 16;

    const float*    Arow = A  + (size_t)(m0 + lrow) * K + lseg;
    const _Float16* Brow = Bt + (size_t)(n0 + lrow) * K + lseg;
    _Float16* da = &sa[lrow * 40 + lseg];
    _Float16* db = &sb[lrow * 40 + lseg];

    // ---- prologue: fetch slab 0 and stage it ----
    v8h  ra0 = cvt8(Arow), ra1 = cvt8(Arow + 8);
    uint4 rb0 = *(const uint4*)Brow, rb1 = *(const uint4*)(Brow + 8);
    *(v8h*)da = ra0;
    *(v8h*)(da + 8) = ra1;
    *(uint4*)db = rb0;
    *(uint4*)(db + 8) = rb1;

    for (int k0 = 32; k0 <= K; k0 += 32) {
        __syncthreads();  // staged slab visible to all waves
        // ---- issue global loads for the next slab (hidden under WMMAs) ----
        if (k0 < K) {
            ra0 = cvt8(Arow + k0);
            ra1 = cvt8(Arow + k0 + 8);
            rb0 = *(const uint4*)(Brow + k0);
            rb1 = *(const uint4*)(Brow + k0 + 8);
        }
        // ---- 8 WMMAs on the current slab ----
        v16h bf[2];
#pragma unroll
        for (int ni = 0; ni < 2; ++ni) {
            const _Float16* base = &sb[(wn * 32 + ni * 16 + l16) * 40];
            bf[ni] = ld_frag(base + kb, base + kb + 16);
        }
#pragma unroll
        for (int mi = 0; mi < 4; ++mi) {
            const _Float16* base = &sa[(wm * 64 + mi * 16 + l16) * 40];
            v16h af = ld_frag(base + kb, base + kb + 16);
            acc[mi][0] = wmma_f16(af, bf[0], acc[mi][0]);
            acc[mi][1] = wmma_f16(af, bf[1], acc[mi][1]);
        }
        __syncthreads();  // all waves done reading before overwrite
        if (k0 < K) {
            *(v8h*)da = ra0;
            *(v8h*)(da + 8) = ra1;
            *(uint4*)db = rb0;
            *(uint4*)(db + 8) = rb1;
        }
    }

    // ---- epilogue: C layout col = lane&15, row = r + 8*(lane>=16) ----
#pragma unroll
    for (int mi = 0; mi < 4; ++mi) {
#pragma unroll
        for (int ni = 0; ni < 2; ++ni) {
            const int gcol = n0 + wn * 32 + ni * 16 + l16;
#pragma unroll
            for (int r = 0; r < 8; ++r) {
                const int grow = m0 + wm * 64 + mi * 16 + r + hs * 8;
                float v = acc[mi][ni][r];
                if (HAS_BIAS) v += bias[gcol];
                if (DO_GELU) v = 0.5f * v * (1.0f + erff(v * 0.70710678118654752f));
                if (DO_RES) v += res[(size_t)grow * N + gcol];
                C[(size_t)grow * N + gcol] = v;
            }
        }
    }
}

// ---------------------------------------------------------------------------
// Flash attention: block = 4 waves (128 thr) = one (b,h) x 64 query rows.
// Each wave owns 16 query rows; key/value blocks of 32 streamed via LDS.
// ---------------------------------------------------------------------------
__global__ __launch_bounds__(128) void attn_kernel(
    const float* __restrict__ Q, const float* __restrict__ K,
    const float* __restrict__ V, float* __restrict__ Y, int T, int D, int H) {
    __shared__ _Float16 sk[32 * 80];      // [key][dh], stride 80 halves
    __shared__ _Float16 sv[64 * 40];      // [dh][key], stride 40 halves
    __shared__ _Float16 sp[4 * 16 * 40];  // per-wave P tile [qrow][key]

    const int t = threadIdx.x;
    const int lane = t & 31, w = t >> 5;
    const int l16 = lane & 15, hs = lane >> 4, kb = hs ? 8 : 0;

    const int nqb = T / 64;
    const int qb = blockIdx.x % nqb;
    const int bh = blockIdx.x / nqb;
    const int b = bh / H, h = bh % H;
    const int q0 = qb * 64;

    // ---- Q fragments (loaded once, pre-scaled by 1/sqrt(64)) ----
    const int qrow = q0 + w * 16 + l16;
    const float* qp = Q + ((size_t)(b * T + qrow)) * D + h * 64;
    v16h aq[2];
#pragma unroll
    for (int c = 0; c < 2; ++c) {
        v16h f;
#pragma unroll
        for (int i = 0; i < 8; ++i) {
            f[i]     = (_Float16)(qp[c * 32 + kb + i] * 0.125f);
            f[8 + i] = (_Float16)(qp[c * 32 + kb + 16 + i] * 0.125f);
        }
        aq[c] = f;
    }

    float mo[8], ls[8];
    v8f o[4];
#pragma unroll
    for (int r = 0; r < 8; ++r) { mo[r] = -1e30f; ls[r] = 0.0f; }
#pragma unroll
    for (int ni = 0; ni < 4; ++ni) o[ni] = vzero8();

    const int jend = q0 + 64;  // causal: block-uniform bound, masked per elem
    for (int j0 = 0; j0 < jend; j0 += 32) {
        __syncthreads();  // all waves done reading previous sk/sv
        // ---- stage K block [32][64] and V block transposed [64][32] ----
        const int krow = t >> 2, seg = (t & 3) * 16;
        const float* kg = K + ((size_t)(b * T + j0 + krow)) * D + h * 64 + seg;
        const float* vg = V + ((size_t)(b * T + j0 + krow)) * D + h * 64 + seg;
        _Float16* skd = &sk[krow * 80 + seg];
        *(v8h*)skd       = cvt8(kg);      // packed 16B LDS stores
        *(v8h*)(skd + 8) = cvt8(kg + 8);
#pragma unroll
        for (int i = 0; i < 16; i += 4) {
            float4 fv = *(const float4*)(vg + i);
            sv[(seg + i + 0) * 40 + krow] = (_Float16)fv.x;
            sv[(seg + i + 1) * 40 + krow] = (_Float16)fv.y;
            sv[(seg + i + 2) * 40 + krow] = (_Float16)fv.z;
            sv[(seg + i + 3) * 40 + krow] = (_Float16)fv.w;
        }
        __syncthreads();

        // ---- S = (Q/8) @ K^T for 32 keys: two 16-wide tiles, K-dim 64 ----
        v8f s0 = vzero8(), s1 = vzero8();
#pragma unroll
        for (int c = 0; c < 2; ++c) {
            const _Float16* b0 = &sk[l16 * 80 + c * 32 + kb];
            const _Float16* b1 = &sk[(16 + l16) * 80 + c * 32 + kb];
            s0 = wmma_f16(aq[c], ld_frag(b0, b0 + 16), s0);
            s1 = wmma_f16(aq[c], ld_frag(b1, b1 + 16), s1);
        }

        // ---- causal mask + online softmax (rows live in 16-lane halves) ----
        const int key0 = j0 + l16, key1 = j0 + 16 + l16;
#pragma unroll
        for (int r = 0; r < 8; ++r) {
            const int qr = q0 + w * 16 + r + hs * 8;
            float x0 = (key0 <= qr) ? s0[r] : -1e30f;
            float x1 = (key1 <= qr) ? s1[r] : -1e30f;
            float cm = fmaxf(x0, x1);
#pragma unroll
            for (int m = 8; m >= 1; m >>= 1) cm = fmaxf(cm, __shfl_xor(cm, m, 32));
            const float mn = fmaxf(mo[r], cm);
            const float al = __expf(mo[r] - mn);
            const float p0 = __expf(x0 - mn);
            const float p1 = __expf(x1 - mn);
            float rs = p0 + p1;
#pragma unroll
            for (int m = 8; m >= 1; m >>= 1) rs += __shfl_xor(rs, m, 32);
            ls[r] = ls[r] * al + rs;
            mo[r] = mn;
#pragma unroll
            for (int ni = 0; ni < 4; ++ni) o[ni][r] *= al;
            const int prow = w * 16 + r + hs * 8;
            sp[prow * 40 + l16]      = (_Float16)p0;
            sp[prow * 40 + 16 + l16] = (_Float16)p1;
        }

        // ---- O += P @ V  (A = P 16x32 from LDS, B = V^T, N = 64) ----
        const _Float16* pb = &sp[(w * 16 + l16) * 40 + kb];
        v16h ap = ld_frag(pb, pb + 16);
#pragma unroll
        for (int ni = 0; ni < 4; ++ni) {
            const _Float16* vb = &sv[(ni * 16 + l16) * 40 + kb];
            o[ni] = wmma_f16(ap, ld_frag(vb, vb + 16), o[ni]);
        }
    }

    // ---- normalize and write y ----
#pragma unroll
    for (int r = 0; r < 8; ++r) {
        const float inv = 1.0f / ls[r];
        const int grow = q0 + w * 16 + r + hs * 8;
        float* yp = Y + ((size_t)(b * T + grow)) * D + h * 64;
#pragma unroll
        for (int ni = 0; ni < 4; ++ni)
            yp[ni * 16 + l16] = o[ni][r] * inv;
    }
}

// ---------------------------------------------------------------------------
// Host orchestration
// ---------------------------------------------------------------------------
static inline char* bump(char*& p, size_t bytes) {
    char* r = p;
    p += (bytes + 255) & ~(size_t)255;
    return r;
}

extern "C" void kernel_launch(void* const* d_in, const int* in_sizes, int n_in,
                              void* d_out, int out_size, void* d_ws, size_t ws_size,
                              hipStream_t stream) {
    constexpr int B = 4, T = 1024, L = 6, D = 1024, H = 16, V = 512, FF = 4 * D;
    constexpr int M = B * T;

    const int*   idx    = (const int*)d_in[0];
    const float* tok    = (const float*)d_in[1];
    const float* pos    = (const float*)d_in[2];
    const float* ln1w   = (const float*)d_in[3];
    const float* ln1b   = (const float*)d_in[4];
    const float* ln2w   = (const float*)d_in[5];
    const float* ln2b   = (const float*)d_in[6];
    const float* wq     = (const float*)d_in[7];
    const float* bq     = (const float*)d_in[8];
    const float* wk     = (const float*)d_in[9];
    const float* bk     = (const float*)d_in[10];
    const float* wv     = (const float*)d_in[11];
    const float* bv     = (const float*)d_in[12];
    const float* wp     = (const float*)d_in[13];
    const float* bp     = (const float*)d_in[14];
    const float* w1     = (const float*)d_in[15];
    const float* b1     = (const float*)d_in[16];
    const float* w2     = (const float*)d_in[17];
    const float* b2     = (const float*)d_in[18];
    const float* lnfw   = (const float*)d_in[19];
    const float* lnfb   = (const float*)d_in[20];
    const float* head_w = (const float*)d_in[21];

    char* p = (char*)d_ws;
    _Float16* wqT = (_Float16*)bump(p, (size_t)L * D * D * 2);
    _Float16* wkT = (_Float16*)bump(p, (size_t)L * D * D * 2);
    _Float16* wvT = (_Float16*)bump(p, (size_t)L * D * D * 2);
    _Float16* wpT = (_Float16*)bump(p, (size_t)L * D * D * 2);
    _Float16* w1T = (_Float16*)bump(p, (size_t)L * D * FF * 2);
    _Float16* w2T = (_Float16*)bump(p, (size_t)L * FF * D * 2);
    _Float16* hT  = (_Float16*)bump(p, (size_t)V * D * 2);
    float* x   = (float*)bump(p, (size_t)M * D * 4);
    float* hb  = (float*)bump(p, (size_t)M * D * 4);
    float* qb2 = (float*)bump(p, (size_t)M * D * 4);
    float* kb2 = (float*)bump(p, (size_t)M * D * 4);
    float* vb2 = (float*)bump(p, (size_t)M * D * 4);
    float* yb  = (float*)bump(p, (size_t)M * D * 4);
    float* mb  = (float*)bump(p, (size_t)M * FF * 4);

    const dim3 tp(32, 8);
    // ---- convert + transpose all weights to f16 [N][K] ----
    for (int l = 0; l < L; ++l) {
        wconv_kernel<<<dim3(D / 32, D / 32), tp, 0, stream>>>(
            wq + (size_t)l * D * D, wqT + (size_t)l * D * D, D, D);
        wconv_kernel<<<dim3(D / 32, D / 32), tp, 0, stream>>>(
            wk + (size_t)l * D * D, wkT + (size_t)l * D * D, D, D);
        wconv_kernel<<<dim3(D / 32, D / 32), tp, 0, stream>>>(
            wv + (size_t)l * D * D, wvT + (size_t)l * D * D, D, D);
        wconv_kernel<<<dim3(D / 32, D / 32), tp, 0, stream>>>(
            wp + (size_t)l * D * D, wpT + (size_t)l * D * D, D, D);
        wconv_kernel<<<dim3(FF / 32, D / 32), tp, 0, stream>>>(
            w1 + (size_t)l * D * FF, w1T + (size_t)l * D * FF, D, FF);
        wconv_kernel<<<dim3(D / 32, FF / 32), tp, 0, stream>>>(
            w2 + (size_t)l * FF * D, w2T + (size_t)l * FF * D, FF, D);
    }
    hconv_kernel<<<(V * D + 255) / 256, 256, 0, stream>>>(head_w, hT, V * D);

    // ---- embedding ----
    embed_kernel<<<M, 256, 0, stream>>>(idx, tok, pos, x, T, D);

    // ---- transformer blocks ----
    for (int l = 0; l < L; ++l) {
        ln_kernel<<<M, 256, 0, stream>>>(x, ln1w + l * D, ln1b + l * D, hb, D);
        gemm_kernel<true, false, false><<<dim3(D / 128, M / 128), 256, 0, stream>>>(
            hb, wqT + (size_t)l * D * D, bq + l * D, nullptr, qb2, M, D, D);
        gemm_kernel<true, false, false><<<dim3(D / 128, M / 128), 256, 0, stream>>>(
            hb, wkT + (size_t)l * D * D, bk + l * D, nullptr, kb2, M, D, D);
        gemm_kernel<true, false, false><<<dim3(D / 128, M / 128), 256, 0, stream>>>(
            hb, wvT + (size_t)l * D * D, bv + l * D, nullptr, vb2, M, D, D);
        attn_kernel<<<B * H * (T / 64), 128, 0, stream>>>(qb2, kb2, vb2, yb, T, D, H);
        gemm_kernel<true, false, true><<<dim3(D / 128, M / 128), 256, 0, stream>>>(
            yb, wpT + (size_t)l * D * D, bp + l * D, x, x, M, D, D);
        ln_kernel<<<M, 256, 0, stream>>>(x, ln2w + l * D, ln2b + l * D, hb, D);
        gemm_kernel<true, true, false><<<dim3(FF / 128, M / 128), 256, 0, stream>>>(
            hb, w1T + (size_t)l * D * FF, b1 + l * FF, nullptr, mb, M, FF, D);
        gemm_kernel<true, false, true><<<dim3(D / 128, M / 128), 256, 0, stream>>>(
            mb, w2T + (size_t)l * FF * D, b2 + l * D, x, x, M, D, FF);
    }

    // ---- final LN + head ----
    ln_kernel<<<M, 256, 0, stream>>>(x, lnfw, lnfb, hb, D);
    gemm_kernel<false, false, false><<<dim3(V / 128, M / 128), 256, 0, stream>>>(
        hb, hT, nullptr, nullptr, (float*)d_out, M, V, D);
}
